// MultiHeadAttention_37529424233060
// MI455X (gfx1250) — matrix operand order repaired and checked by reading, not compiled, not run on hardware
//
#include <hip/hip_runtime.h>

// MI455X / gfx1250: wave32, WMMA bf16 16x16x32, fp32 accumulate.
#define BB 2
#define SS 2048
#define DD 1024
#define HH 16
#define HD 64

typedef __attribute__((ext_vector_type(16))) __bf16 v16bf;
typedef __attribute__((ext_vector_type(8)))  __bf16 v8bf;
typedef __attribute__((ext_vector_type(8)))  float  v8f;

union V16 { v16bf v; v8bf h[2]; };

static __device__ __forceinline__ unsigned short f2bfraw(float f) {
  union { float f; unsigned u; } a; a.f = f;
  unsigned u = a.u;
  unsigned r = (u + 0x7FFFu + ((u >> 16) & 1u)) >> 16;   // RNE
  return (unsigned short)r;
}
static __device__ __forceinline__ __bf16 raw2bf(unsigned short s) {
  union { unsigned short s; __bf16 b; } o; o.s = s; return o.b;
}
static __device__ __forceinline__ v8bf ld8(const unsigned short* p) {
  return *(const v8bf*)p;                                 // global_load_b128
}

// 16-lane-half reductions/broadcast via ds_swizzle (group-of-32 mode)
#define SWZ(v, imm) __int_as_float(__builtin_amdgcn_ds_swizzle(__float_as_int(v), (imm)))

// ---------------------------------------------------------------- converts
__global__ void __launch_bounds__(256)
cvt_f32_to_bf16(const float* __restrict__ in, unsigned short* __restrict__ out, int n) {
  int i = blockIdx.x * blockDim.x + threadIdx.x;
  if (i < n) out[i] = f2bfraw(in[i]);
}
// W[K,N] fp32 -> Wt[N,K] bf16 (so GEMM B-fragments are k-contiguous)
__global__ void __launch_bounds__(256)
cvt_transpose_bf16(const float* __restrict__ in, unsigned short* __restrict__ out,
                   int K, int N) {
  int i = blockIdx.x * blockDim.x + threadIdx.x;
  if (i < K * N) {
    int k = i / N, n = i - k * N;
    out[(size_t)n * K + k] = f2bfraw(in[i]);
  }
}

// ---------------------------------------------------------------- GEMM
// C[M,N] = (A[M,K](bf16) @ Wt[N,K](bf16, pre-transposed) + bias) * oscale.
// One wave computes a 64x32 block = 4x2 WMMA tiles. K-loop step 32.
// GATHERA: A stored [b,h,s,hd] (attention out), logical k = h*64+hd.
// SCATTER: 0 = fp32 row-major, 1 = bf16 [b,h,s,hd], 2 = bf16 [b,h,hd,s].
template <int GATHERA, int SCATTER>
__global__ void __launch_bounds__(256)
gemm_wmma_bf16(const unsigned short* __restrict__ A,
               const unsigned short* __restrict__ Wt,
               const float* __restrict__ bias,
               void* __restrict__ Cout, int M, int N, int K, float oscale) {
  const int lane = threadIdx.x & 31;
  const int wid  = threadIdx.x >> 5;
  const int tid  = blockIdx.x * 8 + wid;
  const int nn   = N >> 5;
  const int tm   = tid / nn;
  const int tn   = tid - tm * nn;
  if (tm >= (M >> 6)) return;
  const int hf  = lane >> 4;
  const int l16 = lane & 15;
  const int m0 = tm << 6, n0 = tn << 5;

  v8f c[4][2];
#pragma unroll
  for (int mi = 0; mi < 4; ++mi)
#pragma unroll
    for (int ni = 0; ni < 2; ++ni)
      c[mi][ni] = (v8f){0.f,0.f,0.f,0.f,0.f,0.f,0.f,0.f};

  for (int k0 = 0; k0 < K; k0 += 32) {
    V16 a[4], b[2];
    const int ka = k0 + (hf << 3);            // lane's first K of run0; run1 = +16
#pragma unroll
    for (int mi = 0; mi < 4; ++mi) {
      const int row = m0 + (mi << 4) + l16;
      size_t i0, i1;
      if (GATHERA) {                           // k = h*64+hd ; 8-runs stay inside hd
        int bb = row >> 11, s = row & (SS - 1);
        int h0 = ka >> 6,        d0 = ka & (HD - 1);
        int h1 = (ka + 16) >> 6, d1 = (ka + 16) & (HD - 1);
        i0 = ((((size_t)bb * HH + h0) * SS + s) << 6) + d0;
        i1 = ((((size_t)bb * HH + h1) * SS + s) << 6) + d1;
      } else {
        i0 = (size_t)row * K + ka;
        i1 = i0 + 16;
      }
      a[mi].h[0] = ld8(A + i0);
      a[mi].h[1] = ld8(A + i1);
    }
#pragma unroll
    for (int ni = 0; ni < 2; ++ni) {
      const unsigned short* p = Wt + (size_t)(n0 + (ni << 4) + l16) * K + k0 + (hf << 3);
      b[ni].h[0] = ld8(p);
      b[ni].h[1] = ld8(p + 16);
    }
#pragma unroll
    for (int mi = 0; mi < 4; ++mi)
#pragma unroll
      for (int ni = 0; ni < 2; ++ni)
        c[mi][ni] = __builtin_amdgcn_wmma_f32_16x16x32_bf16(
            false, a[mi].v, false, b[ni].v, (short)0, c[mi][ni], false, false);
  }

#pragma unroll
  for (int ni = 0; ni < 2; ++ni) {
    const int cn = n0 + (ni << 4) + l16;
    const float bvv = bias[cn];
#pragma unroll
    for (int mi = 0; mi < 4; ++mi) {
#pragma unroll
      for (int i = 0; i < 8; ++i) {
        int cm = m0 + (mi << 4) + (hf << 3) + i;
        float val = (c[mi][ni][i] + bvv) * oscale;
        if (SCATTER == 0) {
          ((float*)Cout)[(size_t)cm * N + cn] = val;
        } else {
          int bb = cm >> 11, s = cm & (SS - 1);
          int hh = cn >> 6,  hd = cn & (HD - 1);
          size_t idx = (SCATTER == 1)
              ? ((((size_t)bb * HH + hh) * SS + s) << 6) + hd       // [b,h,s,hd]
              : (((size_t)bb * HH + hh) * HD + hd) * SS + s;        // [b,h,hd,s]
          ((unsigned short*)Cout)[idx] = f2bfraw(val);
        }
      }
    }
  }
}

// ---------------------------------------------------------------- flash attention
// One wave owns 32 query rows (two 16-row tiles sharing K/V fragments).
// Phase 1: scores+softmax+LDS P-store per tile (K fragments die here).
// Phase 2: both P A-fragments loaded, then each V fragment is loaded right
// before its two PV WMMAs (short live range, shared across tiles).
// MASK=false below the diagonal; exactly one masked diagonal block.
// Q arrives pre-scaled by 1/sqrt(HD) from the Q-projection GEMM.
template <bool MASK>
static __device__ __forceinline__ void attn_block(
    int kk0, int q0, int hf, int l16,
    const unsigned short* __restrict__ Kbh,
    const unsigned short* __restrict__ Vbh,
    unsigned short* __restrict__ pw,
    const V16 (&aq0)[2], const V16 (&aq1)[2], const V16& bones,
    v8f (&acc)[2][4], v8f (&accS)[2], float (&mrun)[2][8]) {
  // ---- K fragments, shared by both query tiles
  V16 k00, k10, k01, k11;          // [key sub-tile 0/1][hd chunk 0/1]
  {
    const unsigned short* kp0 = Kbh + (kk0 + l16) * HD + (hf << 4);
    const unsigned short* kp1 = kp0 + 16 * HD;
    k00.h[0] = ld8(kp0);      k00.h[1] = ld8(kp0 + 8);
    k10.h[0] = ld8(kp1);      k10.h[1] = ld8(kp1 + 8);
    k01.h[0] = ld8(kp0 + 32); k01.h[1] = ld8(kp0 + 40);
    k11.h[0] = ld8(kp1 + 32); k11.h[1] = ld8(kp1 + 40);
  }

  // ---- Phase 1: scores + online softmax + LDS P-store, per query tile
#pragma unroll
  for (int t = 0; t < 2; ++t) {
    v8f s0, s1;
    s0 = __builtin_amdgcn_wmma_f32_16x16x32_bf16(false, aq0[t].v, false, k00.v, (short)0,
                                                 (v8f){0.f,0.f,0.f,0.f,0.f,0.f,0.f,0.f}, false, false);
    s1 = __builtin_amdgcn_wmma_f32_16x16x32_bf16(false, aq0[t].v, false, k10.v, (short)0,
                                                 (v8f){0.f,0.f,0.f,0.f,0.f,0.f,0.f,0.f}, false, false);
    s0 = __builtin_amdgcn_wmma_f32_16x16x32_bf16(false, aq1[t].v, false, k01.v, (short)0, s0, false, false);
    s1 = __builtin_amdgcn_wmma_f32_16x16x32_bf16(false, aq1[t].v, false, k11.v, (short)0, s1, false, false);

#pragma unroll
    for (int i = 0; i < 8; ++i) {
      float v0 = s0[i];
      float v1 = s1[i];
      if (MASK) {
        int qrow = q0 + (t << 4) + (hf << 3) + i;
        v0 = ((kk0 + l16)      <= qrow) ? v0 : -1e30f;
        v1 = ((kk0 + 16 + l16) <= qrow) ? v1 : -1e30f;
      }
      float mx = fmaxf(v0, v1);
      mx = fmaxf(mx, SWZ(mx, 0x041F));
      mx = fmaxf(mx, SWZ(mx, 0x081F));
      mx = fmaxf(mx, SWZ(mx, 0x101F));
      mx = fmaxf(mx, SWZ(mx, 0x201F));         // row max, broadcast in 16-half
      float mnew = fmaxf(mrun[t][i], mx);
      float fsc  = __expf(mrun[t][i] - mnew);
      mrun[t][i] = mnew;
      s0[i] = __expf(v0 - mnew);
      s1[i] = __expf(v1 - mnew);
      acc[t][0][i] *= fsc; acc[t][1][i] *= fsc;
      acc[t][2][i] *= fsc; acc[t][3][i] *= fsc;
      accS[t][i]   *= fsc;                     // running row-sum rescales identically
    }

    unsigned short* pt = pw + (t << 9);
#pragma unroll
    for (int i = 0; i < 8; ++i) {
      int r = (hf << 3) + i;
      pt[r * 32 + l16]      = f2bfraw(s0[i]);
      pt[r * 32 + 16 + l16] = f2bfraw(s1[i]);
    }
  }

  // ---- Phase 2: load both P A-fragments, then short-lived V fragments
  asm volatile("s_wait_dscnt 0x0" ::: "memory");
  V16 pa[2];
#pragma unroll
  for (int t = 0; t < 2; ++t) {
    const unsigned short* pr = pw + (t << 9) + l16 * 32 + (hf << 4);
    pa[t].h[0] = *(const v8bf*)pr;             // ds_load_b128
    pa[t].h[1] = *(const v8bf*)(pr + 8);
  }

#pragma unroll
  for (int j = 0; j < 4; ++j) {
    const unsigned short* vp = Vbh + (size_t)((j << 4) + l16) * SS + kk0 + (hf << 4);
    V16 bv;
    bv.h[0] = ld8(vp);
    bv.h[1] = ld8(vp + 8);
    acc[0][j] = __builtin_amdgcn_wmma_f32_16x16x32_bf16(
        false, pa[0].v, false, bv.v, (short)0, acc[0][j], false, false);
    acc[1][j] = __builtin_amdgcn_wmma_f32_16x16x32_bf16(
        false, pa[1].v, false, bv.v, (short)0, acc[1][j], false, false);
  }
  accS[0] = __builtin_amdgcn_wmma_f32_16x16x32_bf16(
      false, pa[0].v, false, bones.v, (short)0, accS[0], false, false);
  accS[1] = __builtin_amdgcn_wmma_f32_16x16x32_bf16(
      false, pa[1].v, false, bones.v, (short)0, accS[1], false, false);
}

// Q,K: [b,h,s,hd] bf16 (Q pre-scaled).  V: [b,h,hd,s] bf16.  O: [b,h,s,hd] bf16.
__global__ void __launch_bounds__(256)
flash_attn_wmma(const unsigned short* __restrict__ Q,
                const unsigned short* __restrict__ Km,
                const unsigned short* __restrict__ Vt,
                unsigned short* __restrict__ O) {
  __shared__ unsigned short plds[8][1024];      // per-wave 2x(16x32) P tiles (16 KB)
  const int lane = threadIdx.x & 31;
  const int wid  = threadIdx.x >> 5;
  const int gw   = blockIdx.x * 8 + wid;        // 0..2047
  const int qb   = gw & (SS / 32 - 1);          // 32-row query block
  const int bh   = gw >> 6;                     // b*H + h
  const int hf   = lane >> 4;
  const int l16  = lane & 15;
  const int q0   = qb << 5;
  const unsigned short* Qbh = Q  + (size_t)bh * SS * HD;
  const unsigned short* Kbh = Km + (size_t)bh * SS * HD;
  const unsigned short* Vbh = Vt + (size_t)bh * HD * SS;
  unsigned short* pw = plds[wid];

  // Q A-fragments for both 16-row tiles
  V16 aq0[2], aq1[2];
#pragma unroll
  for (int t = 0; t < 2; ++t) {
    const unsigned short* qp = Qbh + (q0 + (t << 4) + l16) * HD + (hf << 4);
    aq0[t].h[0] = ld8(qp);        aq0[t].h[1] = ld8(qp + 8);
    aq1[t].h[0] = ld8(qp + 32);   aq1[t].h[1] = ld8(qp + 40);
  }
  // ones-column B fragment: column 0 = 1.0, all other columns 0
  V16 bones;
  {
    __bf16 ob = raw2bf((l16 == 0) ? (unsigned short)0x3F80 : (unsigned short)0);
#pragma unroll
    for (int h = 0; h < 16; ++h) bones.v[h] = ob;
  }

  v8f acc[2][4];
  v8f accS[2];
  float mrun[2][8];
#pragma unroll
  for (int t = 0; t < 2; ++t) {
#pragma unroll
    for (int j = 0; j < 4; ++j) acc[t][j] = (v8f){0.f,0.f,0.f,0.f,0.f,0.f,0.f,0.f};
    accS[t] = (v8f){0.f,0.f,0.f,0.f,0.f,0.f,0.f,0.f};
#pragma unroll
    for (int i = 0; i < 8; ++i) mrun[t][i] = -1e30f;
  }

  const int nfull = q0 >> 5;               // blocks fully below the diagonal

  for (int kb = 0; kb < nfull; ++kb) {
    const int kk0 = kb << 5;
    // prefetch next key block's K rows (one row per lane) and V keys while WMMAs run
    __builtin_prefetch(Kbh + (kk0 + 32 + lane) * HD, 0, 1);
    __builtin_prefetch(Vbh + (size_t)(lane << 1) * SS + kk0 + 32, 0, 1);
    attn_block<false>(kk0, q0, hf, l16, Kbh, Vbh, pw, aq0, aq1, bones, acc, accS, mrun);
  }
  // exactly one masked diagonal block (keys [q0, q0+32))
  attn_block<true>(q0, q0, hf, l16, Kbh, Vbh, pw, aq0, aq1, bones, acc, accS, mrun);

  // ---- epilogue: O = acc / rowsum ; rowsum lives in column 0 of accS
  unsigned short* Obh = O + (size_t)bh * SS * HD;
#pragma unroll
  for (int t = 0; t < 2; ++t) {
#pragma unroll
    for (int i = 0; i < 8; ++i) {
      int qrow = q0 + (t << 4) + (hf << 3) + i;
      float ls  = SWZ(accS[t][i], 0x0010);     // broadcast lane (lane&16) -> 16-half
      float inv = 1.0f / ls;
#pragma unroll
      for (int j = 0; j < 4; ++j)
        Obh[qrow * HD + (j << 4) + l16] = f2bfraw(acc[t][j][i] * inv);
    }
  }
}

// ---------------------------------------------------------------- launch
extern "C" void kernel_launch(void* const* d_in, const int* in_sizes, int n_in,
                              void* d_out, int out_size, void* d_ws, size_t ws_size,
                              hipStream_t stream) {
  const float* X  = (const float*)d_in[0];
  const float* Wq = (const float*)d_in[1];
  const float* bq = (const float*)d_in[2];
  const float* Wk = (const float*)d_in[3];
  const float* bk = (const float*)d_in[4];
  const float* Wv = (const float*)d_in[5];
  const float* bv = (const float*)d_in[6];
  const float* Wo = (const float*)d_in[7];
  const float* bo = (const float*)d_in[8];

  char* ws = (char*)d_ws;
  const size_t MB = 1u << 20;
  unsigned short* Xbf  = (unsigned short*)(ws + 0 * MB);   // [4096,1024] row-major
  unsigned short* Wqbf = (unsigned short*)(ws + 8 * MB);   // [N,K] transposed
  unsigned short* Wkbf = (unsigned short*)(ws + 10 * MB);
  unsigned short* Wvbf = (unsigned short*)(ws + 12 * MB);
  unsigned short* Wobf = (unsigned short*)(ws + 14 * MB);
  unsigned short* Qbf  = (unsigned short*)(ws + 16 * MB);  // [b,h,s,hd] (pre-scaled)
  unsigned short* Kbf  = (unsigned short*)(ws + 24 * MB);  // [b,h,s,hd]
  unsigned short* Vbf  = (unsigned short*)(ws + 32 * MB);  // [b,h,hd,s]
  unsigned short* Obf  = (unsigned short*)(ws + 40 * MB);  // [b,h,s,hd]

  const int NX = BB * SS * DD;   // 4M
  const int NW = DD * DD;        // 1M
  cvt_f32_to_bf16<<<NX / 256, 256, 0, stream>>>(X, Xbf, NX);
  cvt_transpose_bf16<<<NW / 256, 256, 0, stream>>>(Wq, Wqbf, DD, DD);
  cvt_transpose_bf16<<<NW / 256, 256, 0, stream>>>(Wk, Wkbf, DD, DD);
  cvt_transpose_bf16<<<NW / 256, 256, 0, stream>>>(Wv, Wvbf, DD, DD);
  cvt_transpose_bf16<<<NW / 256, 256, 0, stream>>>(Wo, Wobf, DD, DD);

  const int M = BB * SS;                                    // 4096
  const int gemmBlocks = (M / 64) * (DD / 32) / 8;          // 256 blocks x 8 waves
  // Q pre-scaled by 1/sqrt(HD) so attention needs no score scaling.
  gemm_wmma_bf16<0, 1><<<gemmBlocks, 256, 0, stream>>>(Xbf, Wqbf, bq, Qbf, M, DD, DD, 0.125f);
  gemm_wmma_bf16<0, 1><<<gemmBlocks, 256, 0, stream>>>(Xbf, Wkbf, bk, Kbf, M, DD, DD, 1.0f);
  gemm_wmma_bf16<0, 2><<<gemmBlocks, 256, 0, stream>>>(Xbf, Wvbf, bv, Vbf, M, DD, DD, 1.0f);

  const int attnBlocks = BB * HH * (SS / 32) / 8;           // 256
  flash_attn_wmma<<<attnBlocks, 256, 0, stream>>>(Qbf, Kbf, Vbf, Obf);

  gemm_wmma_bf16<1, 0><<<gemmBlocks, 256, 0, stream>>>(Obf, Wobf, bo, (float*)d_out,
                                                       M, DD, DD, 1.0f);
  (void)in_sizes; (void)n_in; (void)out_size; (void)ws_size;
}
